// MultiHeadAttentionLayer_48876727828589
// MI455X (gfx1250) — matrix operand order repaired and checked by reading, not compiled, not run on hardware
//
#include <hip/hip_runtime.h>

// ---------------------------------------------------------------------------
// MultiHeadAttention forward for MI455X (gfx1250, wave32, WMMA bf16->f32)
//   B=2, S=2048, D=1024, H=16, Hd=64
//   d_out = [ x : B*S*D f32 | attention : B*H*S*S f32 ]
//   d_ws  = [ Q bf16 8MB | K bf16 8MB | V bf16 8MB | Ctx bf16 8MB ]
// ---------------------------------------------------------------------------

typedef __attribute__((ext_vector_type(16))) __bf16    v16bf;
typedef __attribute__((ext_vector_type(8)))  float     v8f;
typedef __attribute__((ext_vector_type(4)))  unsigned  u32x4;
typedef __attribute__((ext_vector_type(8)))  unsigned  u32x8;

#define S_LEN   2048
#define D_DIM   1024
#define NHEAD   16
#define HDIM    64
#define BS_ROWS 4096   // B*S

// A-matrix (16x32 bf16) K-pair base for VGPR i, lane-half h (contiguous: k = 16h + 2i)
__device__ __forceinline__ int a_kbase(int i, int h) {
    return h * 16 + 2 * i;
}

union FragU { v16bf v; unsigned u[8]; __bf16 h[16]; };

// ---------------------------------------------------------------------------
// Kernel 1: projection GEMM  Out[b,h,s,hd] = (In[BS,D] @ W[D,D]) head-split
// block = 256 threads (8 waves), tile 64x64, K-step 32
// ---------------------------------------------------------------------------
__global__ __launch_bounds__(256) void proj_kernel(const float* __restrict__ In,
                                                   const float* __restrict__ W,
                                                   __bf16* __restrict__ Out) {
    __shared__ __bf16 As[64][32];   // row-major [m][k]
    __shared__ __bf16 Bs[64][32];   // transposed [n][k] so K is contiguous

    const int tid  = threadIdx.x;
    const int lane = tid & 31;
    const int wave = tid >> 5;
    const int wm   = wave & 1;      // 0..1 -> 32-row half
    const int wn   = wave >> 1;     // 0..3 -> 16-col strip
    const int m0   = blockIdx.y * 64;
    const int n0   = blockIdx.x * 64;
    const int half = lane >> 4;
    const int lr   = lane & 15;

    v8f acc0 = {}; v8f acc1 = {};

    for (int k0 = 0; k0 < D_DIM; k0 += 32) {
        __syncthreads();
        // prefetch next K-step tiles into L2/L0 while this step computes
        if (k0 + 32 < D_DIM) {
            __builtin_prefetch(In + (size_t)(m0 + (tid >> 3)) * D_DIM + k0 + 32, 0, 1);
            __builtin_prefetch(W + (size_t)(k0 + 32 + (tid >> 4)) * D_DIM + n0, 0, 1);
        }
        // stage A tile: 64x32 f32 -> bf16   (512 float4 slots, 2 per thread)
        #pragma unroll
        for (int s = 0; s < 2; ++s) {
            int slot = tid + s * 256;
            int mm = slot >> 3;
            int kk = (slot & 7) << 2;
            float4 f = *(const float4*)(In + (size_t)(m0 + mm) * D_DIM + k0 + kk);
            __bf16* d = &As[mm][kk];
            d[0] = (__bf16)f.x; d[1] = (__bf16)f.y; d[2] = (__bf16)f.z; d[3] = (__bf16)f.w;
        }
        // stage B tile transposed: W[k0+kk][n0+nn] -> Bs[nn][kk]
        #pragma unroll
        for (int s = 0; s < 2; ++s) {
            int slot = tid + s * 256;
            int kk = slot >> 4;
            int nn = (slot & 15) << 2;
            float4 f = *(const float4*)(W + (size_t)(k0 + kk) * D_DIM + n0 + nn);
            Bs[nn + 0][kk] = (__bf16)f.x; Bs[nn + 1][kk] = (__bf16)f.y;
            Bs[nn + 2][kk] = (__bf16)f.z; Bs[nn + 3][kk] = (__bf16)f.w;
        }
        __syncthreads();

        FragU b, a0, a1;
        #pragma unroll
        for (int j = 0; j < 8; ++j) {
            int kb = half * 16 + 2 * j;
            b.u[j] = *(const unsigned*)&Bs[wn * 16 + lr][kb];
        }
        #pragma unroll
        for (int i = 0; i < 8; ++i) {
            int kb = a_kbase(i, half);
            a0.u[i] = *(const unsigned*)&As[wm * 32 + lr][kb];
            a1.u[i] = *(const unsigned*)&As[wm * 32 + 16 + lr][kb];
        }
        acc0 = __builtin_amdgcn_wmma_f32_16x16x32_bf16(false, a0.v, false, b.v,
                                                       (short)0, acc0, false, false);
        acc1 = __builtin_amdgcn_wmma_f32_16x16x32_bf16(false, a1.v, false, b.v,
                                                       (short)0, acc1, false, false);
    }

    // epilogue: bf16 store in head-split [b][h][s][hd]
    #pragma unroll
    for (int sub = 0; sub < 2; ++sub) {
        v8f acc = sub ? acc1 : acc0;
        #pragma unroll
        for (int r = 0; r < 8; ++r) {
            int gm = m0 + wm * 32 + sub * 16 + r + 8 * half;
            int gn = n0 + wn * 16 + lr;
            int bb = gm >> 11, ss = gm & (S_LEN - 1);
            int hh = gn >> 6,  hd = gn & (HDIM - 1);
            Out[((((size_t)bb * NHEAD + hh) * S_LEN + ss) * HDIM) + hd] = (__bf16)acc[r];
        }
    }
}

// ---------------------------------------------------------------------------
// Kernel 2: fused attention for one (b, h, 16-query block)
//   Q tile via Tensor Data Mover (TDM) -> scores (WMMA) -> softmax in LDS ->
//   attn f32 to d_out -> P@V (WMMA + ds_add_f32 reduce) -> bf16 context
// dynamic LDS: Sc 16x2048 f32 | Qs 16x64 bf16 | red 16x16 f32 | stat 16 f32 |
//              Ctx 16x64 f32
// ---------------------------------------------------------------------------
__global__ __launch_bounds__(256) void attn_kernel(const __bf16* __restrict__ Qw,
                                                   const __bf16* __restrict__ Kw,
                                                   const __bf16* __restrict__ Vw,
                                                   float* __restrict__ attn,
                                                   __bf16* __restrict__ Ctxw) {
    extern __shared__ char smem[];
    float*  Sc   = (float*)smem;                               // 128 KB
    __bf16* Qs   = (__bf16*)(smem + 16 * S_LEN * 4);           // 2 KB
    float*  red  = (float*)(smem + 16 * S_LEN * 4 + 2048);     // 1 KB
    float*  stat = red + 256;                                  // 64 B
    float*  Ctx  = stat + 16;                                  // 4 KB

    const int tid  = threadIdx.x;
    const int lane = tid & 31;
    const int wave = tid >> 5;
    const int half = lane >> 4;
    const int lr   = lane & 15;
    const int qb   = blockIdx.x;        // 0..127
    const int hy   = blockIdx.y;        // head
    const int bz   = blockIdx.z;        // batch
    const int q0   = qb * 16;

    const size_t headBase = ((size_t)bz * NHEAD + hy) * S_LEN;
    const __bf16* Qh = Qw + (headBase + q0) * HDIM;
    const __bf16* Kh = Kw + headBase * HDIM;
    const __bf16* Vh = Vw + headBase * HDIM;

    // ---- load Q tile (16x64 bf16 = 2KB) via Tensor Data Mover ----
    // D# group0: count=1 | lds_addr | global_addr | type=2
    // D# group1: data_size=2B, tensor_dim0=64, tensor_dim1=2048,
    //            tile_dim0=64, tile_dim1=16, tensor_dim0_stride=64
    if (wave == 0) {
        unsigned long long ga = (unsigned long long)(uintptr_t)Qh;
        unsigned lds = (unsigned)(uintptr_t)Qs;     // flat addr[31:0] == LDS offset
        u32x4 g0;
        g0.x = 1u;                                           // count=1 (valid user D#)
        g0.y = lds;                                          // lds_addr
        g0.z = (unsigned)ga;                                 // global_addr[31:0]
        g0.w = (unsigned)((ga >> 32) & 0x1FFFFFFull) | (2u << 30); // ga[56:32]|type=2
        u32x8 g1;
        g1.s0 = 1u << 16;                                    // data_size = 2 bytes
        g1.s1 = (unsigned)HDIM << 16;                        // tensor_dim0 = 64
        g1.s2 = (unsigned)S_LEN << 16;                       // tensor_dim1 = 2048
        g1.s3 = (unsigned)HDIM << 16;                        // tile_dim0 = 64
        g1.s4 = 16u;                                         // tile_dim1 = 16
        g1.s5 = (unsigned)HDIM;                              // tensor_dim0_stride = 64
        g1.s6 = 0u;
        g1.s7 = 0u;
        asm volatile("tensor_load_to_lds %0, %1" :: "s"(g0), "s"(g1) : "memory");
        __builtin_amdgcn_s_wait_tensorcnt(0);
    }
    __syncthreads();

    // A fragments of Q: k = hd 0..31 and 32..63 (fixed for whole block)
    FragU qa0, qa1;
    #pragma unroll
    for (int i = 0; i < 8; ++i) {
        int kb = a_kbase(i, half);
        qa0.u[i] = *(const unsigned*)&Qs[lr * HDIM + kb];
        qa1.u[i] = *(const unsigned*)&Qs[lr * HDIM + 32 + kb];
    }

    // ---- phase 1: scores = (Q K^T) * 1/sqrt(Hd), 16 key-tiles per wave ----
    for (int kt = wave; kt < S_LEN / 16; kt += 8) {
        int s0 = kt * 16;
        const __bf16* kp = Kh + (size_t)(s0 + lr) * HDIM + half * 16;
        FragU b0, b1;
        #pragma unroll
        for (int j = 0; j < 8; ++j) {
            b0.u[j] = *(const unsigned*)(kp + 2 * j);        // hd 0..31
            b1.u[j] = *(const unsigned*)(kp + 32 + 2 * j);   // hd 32..63
        }
        v8f acc = {};
        acc = __builtin_amdgcn_wmma_f32_16x16x32_bf16(false, qa0.v, false, b0.v,
                                                      (short)0, acc, false, false);
        acc = __builtin_amdgcn_wmma_f32_16x16x32_bf16(false, qa1.v, false, b1.v,
                                                      (short)0, acc, false, false);
        #pragma unroll
        for (int r = 0; r < 8; ++r)
            Sc[(r + 8 * half) * S_LEN + s0 + lr] = acc[r] * 0.125f; // 1/sqrt(64)
    }
    __syncthreads();

    // ---- phase 2: softmax over each row of 2048, write attn f32 ----
    {
        const int r = tid >> 4;
        const int c = tid & 15;
        float* row = Sc + r * S_LEN;
        float mx = -3.0e38f;
        for (int j = c; j < S_LEN; j += 16) mx = fmaxf(mx, row[j]);
        red[r * 16 + c] = mx;
        __syncthreads();
        if (c == 0) {
            float m = red[r * 16];
            #pragma unroll
            for (int t = 1; t < 16; ++t) m = fmaxf(m, red[r * 16 + t]);
            stat[r] = m;
        }
        __syncthreads();
        float m = stat[r];
        float sum = 0.f;
        for (int j = c; j < S_LEN; j += 16) {
            float e = __expf(row[j] - m);
            row[j] = e;
            sum += e;
        }
        red[r * 16 + c] = sum;
        __syncthreads();
        if (c == 0) {
            float s = 0.f;
            #pragma unroll
            for (int t = 0; t < 16; ++t) s += red[r * 16 + t];
            stat[r] = 1.0f / s;
        }
        __syncthreads();
        float inv = stat[r];
        float* aout = attn + (headBase + q0 + r) * S_LEN;
        for (int j = c; j < S_LEN; j += 16) {
            float p = row[j] * inv;
            row[j]  = p;       // keep normalized probs for P@V
            aout[j] = p;       // materialize attention (f32)
        }
    }

    // ---- phase 3: context = P @ V  (each wave owns 8 K-chunks of 32) ----
    for (int i = tid; i < 16 * HDIM; i += 256) Ctx[i] = 0.f;
    __syncthreads();

    v8f cacc[4] = {};
    for (int ch = wave; ch < S_LEN / 32; ch += 8) {
        int kb = ch * 32;
        FragU a;
        #pragma unroll
        for (int i = 0; i < 8; ++i) {
            float2 f = *(const float2*)&Sc[lr * S_LEN + kb + a_kbase(i, half)];
            a.h[2 * i]     = (__bf16)f.x;
            a.h[2 * i + 1] = (__bf16)f.y;
        }
        #pragma unroll
        for (int t = 0; t < 4; ++t) {
            FragU bv;
            int hd = t * 16 + lr;
            #pragma unroll
            for (int j = 0; j < 8; ++j) {
                int kk = kb + half * 16 + 2 * j;
                bv.h[2 * j]     = Vh[(size_t)kk * HDIM + hd];
                bv.h[2 * j + 1] = Vh[(size_t)(kk + 1) * HDIM + hd];
            }
            cacc[t] = __builtin_amdgcn_wmma_f32_16x16x32_bf16(false, a.v, false, bv.v,
                                                              (short)0, cacc[t], false, false);
        }
    }
    #pragma unroll
    for (int t = 0; t < 4; ++t)
        #pragma unroll
        for (int r = 0; r < 8; ++r)
            atomicAdd(&Ctx[(r + 8 * half) * HDIM + t * 16 + lr], cacc[t][r]);
    __syncthreads();

    // write context as bf16 into [B*S, D] (merge-heads layout)
    for (int i = tid; i < 16 * HDIM; i += 256) {
        int rr = i >> 6, hd = i & (HDIM - 1);
        size_t row = (size_t)bz * S_LEN + q0 + rr;
        Ctxw[row * D_DIM + hy * HDIM + hd] = (__bf16)Ctx[i];
    }
}

// ---------------------------------------------------------------------------
// Kernel 3: output projection  x = Ctx[BS,D](bf16) @ Wo[D,D] -> f32
// ---------------------------------------------------------------------------
__global__ __launch_bounds__(256) void outproj_kernel(const __bf16* __restrict__ X,
                                                      const float* __restrict__ W,
                                                      float* __restrict__ Out) {
    __shared__ __bf16 As[64][32];
    __shared__ __bf16 Bs[64][32];   // [n][k]

    const int tid  = threadIdx.x;
    const int lane = tid & 31;
    const int wave = tid >> 5;
    const int wm   = wave & 1;
    const int wn   = wave >> 1;
    const int m0   = blockIdx.y * 64;
    const int n0   = blockIdx.x * 64;
    const int half = lane >> 4;
    const int lr   = lane & 15;

    v8f acc0 = {}; v8f acc1 = {};

    for (int k0 = 0; k0 < D_DIM; k0 += 32) {
        __syncthreads();
        if (k0 + 32 < D_DIM) {
            __builtin_prefetch(X + (size_t)(m0 + (tid >> 2)) * D_DIM + k0 + 32, 0, 1);
            __builtin_prefetch(W + (size_t)(k0 + 32 + (tid >> 4)) * D_DIM + n0, 0, 1);
        }
        // A: already bf16, 64x32 = 4KB -> one uint4 (8 bf16) per thread
        {
            int mm = tid >> 2;
            int kk = (tid & 3) << 3;
            *(uint4*)&As[mm][kk] =
                *(const uint4*)(X + (size_t)(m0 + mm) * D_DIM + k0 + kk);
        }
        // B: f32 -> bf16 transposed
        #pragma unroll
        for (int s = 0; s < 2; ++s) {
            int slot = tid + s * 256;
            int kk = slot >> 4;
            int nn = (slot & 15) << 2;
            float4 f = *(const float4*)(W + (size_t)(k0 + kk) * D_DIM + n0 + nn);
            Bs[nn + 0][kk] = (__bf16)f.x; Bs[nn + 1][kk] = (__bf16)f.y;
            Bs[nn + 2][kk] = (__bf16)f.z; Bs[nn + 3][kk] = (__bf16)f.w;
        }
        __syncthreads();

        FragU b, a0, a1;
        #pragma unroll
        for (int j = 0; j < 8; ++j) {
            int kb = half * 16 + 2 * j;
            b.u[j] = *(const unsigned*)&Bs[wn * 16 + lr][kb];
        }
        #pragma unroll
        for (int i = 0; i < 8; ++i) {
            int kb = a_kbase(i, half);
            a0.u[i] = *(const unsigned*)&As[wm * 32 + lr][kb];
            a1.u[i] = *(const unsigned*)&As[wm * 32 + 16 + lr][kb];
        }
        acc0 = __builtin_amdgcn_wmma_f32_16x16x32_bf16(false, a0.v, false, b.v,
                                                       (short)0, acc0, false, false);
        acc1 = __builtin_amdgcn_wmma_f32_16x16x32_bf16(false, a1.v, false, b.v,
                                                       (short)0, acc1, false, false);
    }

    #pragma unroll
    for (int sub = 0; sub < 2; ++sub) {
        v8f acc = sub ? acc1 : acc0;
        #pragma unroll
        for (int r = 0; r < 8; ++r) {
            int gm = m0 + wm * 32 + sub * 16 + r + 8 * half;
            int gn = n0 + wn * 16 + lr;
            Out[(size_t)gm * D_DIM + gn] = acc[r];
        }
    }
}

// ---------------------------------------------------------------------------
extern "C" void kernel_launch(void* const* d_in, const int* in_sizes, int n_in,
                              void* d_out, int out_size, void* d_ws, size_t ws_size,
                              hipStream_t stream) {
    (void)in_sizes; (void)n_in; (void)out_size; (void)ws_size;

    const float* q  = (const float*)d_in[0];
    const float* k  = (const float*)d_in[1];
    const float* v  = (const float*)d_in[2];
    const float* wq = (const float*)d_in[3];
    const float* wk = (const float*)d_in[4];
    const float* wv = (const float*)d_in[5];
    const float* wo = (const float*)d_in[6];

    float* out_x    = (float*)d_out;                                   // [B,S,D]
    float* out_attn = out_x + (size_t)BS_ROWS * D_DIM;                 // [B,H,S,S]

    const size_t elems = (size_t)BS_ROWS * D_DIM;   // 4,194,304
    __bf16* Qw = (__bf16*)d_ws;
    __bf16* Kw = Qw + elems;
    __bf16* Vw = Kw + elems;
    __bf16* Cw = Vw + elems;

    dim3 gGemm(D_DIM / 64, BS_ROWS / 64);   // (16, 64)
    proj_kernel<<<gGemm, 256, 0, stream>>>(q, wq, Qw);
    proj_kernel<<<gGemm, 256, 0, stream>>>(k, wk, Kw);
    proj_kernel<<<gGemm, 256, 0, stream>>>(v, wv, Vw);

    size_t smem = (size_t)16 * S_LEN * 4      // Sc
                + (size_t)16 * HDIM * 2       // Qs
                + 256 * 4                     // red
                + 16 * 4                      // stat
                + (size_t)16 * HDIM * 4;      // Ctx  => 138,304 B (< 320KB WGP LDS)
    dim3 gAttn(S_LEN / 16, NHEAD, 2);         // (128, 16, 2)
    attn_kernel<<<gAttn, 256, smem, stream>>>(Qw, Kw, Vw, out_attn, Cw);

    outproj_kernel<<<gGemm, 256, 0, stream>>>(Cw, wo, out_x);
}
